// ColumnGraphMemory_59794534695164
// MI455X (gfx1250) — matrix operand order repaired and checked by reading, not compiled, not run on hardware
//
#include <hip/hip_runtime.h>
#include <hip/hip_bf16.h>

// ---------------- problem constants ----------------
constexpr int B_  = 2;
constexpr int N_  = 16384;
constexpr int K_  = 16;
constexpr int DS  = 128;     // D_s
constexpr int DID = 64;      // D_id
constexpr int XW  = DS + DID;        // 192
constexpr int UW  = 2 * DS + DID;    // 320
constexpr int HID = 256;     // Hc == Hq == Hu
constexpr int HD  = 128;     // H*Dq

typedef __attribute__((ext_vector_type(16))) __bf16 v16bf;
typedef __attribute__((ext_vector_type(8)))  float  v8f;

// ---------------- WMMA core ----------------
__device__ inline v8f wmma_bf16(v16bf a, v16bf b, v8f c) {
  return __builtin_amdgcn_wmma_f32_16x16x32_bf16(
      /*neg_a=*/false, a, /*neg_b=*/false, b,
      /*c_mod=*/(short)0, c, /*reuse_a=*/false, /*reuse_b=*/false);
}

// Fragment buffers hold, per 16x32 (A) or 32x16 (B) tile, 32 lanes x 16 bf16
// contiguous (32B per lane) so a lane's whole fragment is one v16bf load
// (lowers to 2x b128). Tile t, lane l -> elems [ (t*32+l)*16 , +16 ).
__device__ inline v16bf load_frag(const __bf16* buf, int tile, int lane) {
  return ((const v16bf*)buf)[tile * 32 + lane];
}

// Inverse of the A-matrix 16x32 bf16 register layout (ISA 7.12.2):
// element (row m, k) lives at lane h*16+m, vec elem 2*(g*4+p)+c where
// k%32 = g*16 + h*8 + p*2 + c.
__device__ inline int afrag_idx(int m, int k) {
  int kt = k >> 5, kk = k & 31;
  int g = kk >> 4;
  int r = kk & 15;
  int h = r >> 3;
  int r2 = r & 7;
  int lane = h * 16 + m;
  int e = 2 * (g * 4 + (r2 >> 1)) + (r2 & 1);
  return (kt * 32 + lane) * 16 + e;
}

// branch-free GELU (tanh form) and sigmoid via v_exp_f32
__device__ inline float gelu_tanh(float x) {
  float u = 0.7978845608f * (x + 0.044715f * x * x * x);
  float t = 1.0f - 2.0f / (1.0f + __expf(2.0f * u));   // tanh(u)
  return 0.5f * x * (1.0f + t);
}
__device__ inline float sigmoidf(float x) { return 1.0f / (1.0f + __expf(-x)); }

// ---------------- kernel 0: weight swizzle fp32 -> bf16 B-fragments --------
// B-matrix 32x16 layout: elem e holds K = (e>>3)*16 + h*8 + (e&7) at N = lane&15.
__device__ inline void swizzleB(const float* w, __bf16* dst, int Kdim, int Ncols,
                                int tid, int stride) {
  int total = Kdim * Ncols;
  int ktiles = Kdim >> 5;
  for (int o = tid; o < total; o += stride) {
    int e = o & 15, lane = (o >> 4) & 31, tile = o >> 9;
    int kt = tile % ktiles, nt = tile / ktiles;
    int h = lane >> 4, n = lane & 15;
    int k = kt * 32 + (e >> 3) * 16 + h * 8 + (e & 7);
    dst[o] = (__bf16)w[(size_t)k * Ncols + nt * 16 + n];
  }
}

__global__ void prep_kernel(const float* c1, const float* q1, const float* k1,
                            const float* u1, const float* c2, const float* q2,
                            const float* k2, const float* u2, const float* cid,
                            __bf16* dc1, __bf16* dq1, __bf16* dk1, __bf16* du1,
                            __bf16* dc2, __bf16* dq2, __bf16* dk2, __bf16* du2,
                            __bf16* dcid) {
  int i = blockIdx.x * blockDim.x + threadIdx.x;
  int st = gridDim.x * blockDim.x;
  swizzleB(c1, dc1, XW, HID, i, st);
  swizzleB(q1, dq1, XW, HID, i, st);
  swizzleB(k1, dk1, DID, HID, i, st);
  swizzleB(u1, du1, UW, HID, i, st);
  swizzleB(c2, dc2, HID, HD, i, st);
  swizzleB(q2, dq2, HID, HD, i, st);
  swizzleB(k2, dk2, HID, HD, i, st);
  swizzleB(u2, du2, HID, HD, i, st);
  for (int t = i; t < N_ * DID; t += st) dcid[t] = (__bf16)cid[t];
}

// ---------------- kernel 1: k projection + decay (B-independent) ----------
__global__ __launch_bounds__(256) void kproj_kernel(
    const __bf16* cidb, const float* cidf,
    const __bf16* k1w, const float* k1b,
    const __bf16* k2w, const float* k2b,
    const float* decw, const float* decb,
    float* kvec, float* decay) {
  __shared__ __align__(32) __bf16 A[16 * DID];     // A-fragment layout, 2 ktiles
  __shared__ __align__(32) __bf16 Hh[16 * HID];    // A-fragment layout, 8 ktiles
  int tid = threadIdx.x, wid = tid >> 5, lane = tid & 31;
  int row0 = blockIdx.x * 16;

  for (int t = tid; t < 16 * DID; t += 256) {
    int m = t >> 6, d = t & 63;
    A[afrag_idx(m, d)] = cidb[(size_t)(row0 + m) * DID + d];
  }
  __syncthreads();

  // layer1: 64 -> 256, 16 column tiles, 2 per wave, 2 K-steps each
#pragma unroll
  for (int tt = 0; tt < 2; ++tt) {
    int nt = wid * 2 + tt;
    int n0 = nt * 16;
    v8f acc = {};
#pragma unroll
    for (int ks = 0; ks < 2; ++ks)
      acc = wmma_bf16(load_frag(A, ks, lane), load_frag(k1w, nt * 2 + ks, lane), acc);
    int n = lane & 15, h = lane >> 4;
    float bias = k1b[n0 + n];
#pragma unroll
    for (int r = 0; r < 8; ++r)
      Hh[afrag_idx(r + h * 8, n0 + n)] = (__bf16)gelu_tanh(acc[r] + bias);
  }
  __syncthreads();

  // layer2: 256 -> 128, one 16-col tile per wave, 8 K-steps
  {
    int nt = wid;
    int n0 = nt * 16;
    v8f acc = {};
#pragma unroll
    for (int ks = 0; ks < 8; ++ks)
      acc = wmma_bf16(load_frag(Hh, ks, lane), load_frag(k2w, nt * 8 + ks, lane), acc);
    int n = lane & 15, h = lane >> 4;
    float bias = k2b[n0 + n];
#pragma unroll
    for (int r = 0; r < 8; ++r)
      kvec[(size_t)(row0 + r + h * 8) * HD + n0 + n] = acc[r] + bias;
  }

  if (tid < 16) {
    float acc = decb[0];
    const float* cr = cidf + (size_t)(row0 + tid) * DID;
    for (int d = 0; d < DID; ++d) acc += cr[d] * decw[d];
    decay[row0 + tid] = sigmoidf(acc);
  }
}

// ---------------- kernel 2: norm + content/q MLPs + edge score + scatter ---
__global__ __launch_bounds__(256) void fused_main_kernel(
    const float* s, const int* nbrs, const float* ebias, const __bf16* cidb,
    const float* cnw,
    const __bf16* c1w, const float* c1b, const __bf16* c2w, const float* c2b,
    const __bf16* q1w, const float* q1b, const __bf16* q2w, const float* q2b,
    const float* kvec, float* sn, float* incoming) {
  __shared__ float  sf[16 * DS];
  __shared__ float  rstd[16];
  __shared__ __align__(32) __bf16 X[16 * XW];      // A-frag, 6 ktiles
  __shared__ __align__(32) __bf16 Hc[16 * HID];    // A-frag, 8 ktiles
  __shared__ __align__(32) __bf16 Hq[16 * HID];    // A-frag, 8 ktiles
  __shared__ float  Mo[16 * DS];
  __shared__ float  Qo[16 * HD];
  int tid = threadIdx.x, wid = tid >> 5, lane = tid & 31;
  int row0 = blockIdx.x * 16;
  int b = blockIdx.y;
  const float* sB = s + (size_t)b * N_ * DS;

  for (int t = tid; t < 16 * DS; t += 256) sf[t] = sB[(size_t)row0 * DS + t];
  __syncthreads();
  if (tid < 16) {
    float ss = 0.f;
    const float* r = sf + tid * DS;
    for (int d = 0; d < DS; ++d) ss += r[d] * r[d];
    rstd[tid] = rsqrtf(ss / (float)DS + 1e-6f);
  }
  __syncthreads();

  // x = [rmsnorm(s)*g , col_id] written straight into A-fragment layout
  for (int t = tid; t < 16 * XW; t += 256) {
    int m = t / XW, d = t % XW;
    float v;
    if (d < DS) {
      v = sf[m * DS + d] * rstd[m] * cnw[d];
      sn[((size_t)b * N_ + row0 + m) * DS + d] = v;
    } else {
      v = (float)cidb[(size_t)(row0 + m) * DID + (d - DS)];
    }
    X[afrag_idx(m, d)] = (__bf16)v;
  }
  __syncthreads();

  // layer1 of content & q MLPs fused: 32 column tiles total, 4 per wave
#pragma unroll
  for (int tt = 0; tt < 4; ++tt) {
    int t = wid * 4 + tt;
    bool isq = t >= 16;
    int nt = isq ? t - 16 : t;
    int n0 = nt * 16;
    const __bf16* w   = isq ? q1w : c1w;
    const float*  bb  = isq ? q1b : c1b;
    __bf16*       dst = isq ? Hq : Hc;
    v8f acc = {};
#pragma unroll
    for (int ks = 0; ks < 6; ++ks)
      acc = wmma_bf16(load_frag(X, ks, lane), load_frag(w, nt * 6 + ks, lane), acc);
    int n = lane & 15, h = lane >> 4;
    float bias = bb[n0 + n];
#pragma unroll
    for (int r = 0; r < 8; ++r)
      dst[afrag_idx(r + h * 8, n0 + n)] = (__bf16)gelu_tanh(acc[r] + bias);
  }
  __syncthreads();

  // layer2: m_out (8 tiles) + q (8 tiles), 2 tiles per wave, 8 K-steps
#pragma unroll
  for (int tt = 0; tt < 2; ++tt) {
    int t = wid * 2 + tt;
    bool isq = t >= 8;
    int nt = isq ? t - 8 : t;
    int n0 = nt * 16;
    const __bf16* Asrc = isq ? Hq : Hc;
    const __bf16* w    = isq ? q2w : c2w;
    const float*  bb   = isq ? q2b : c2b;
    float*        dst  = isq ? Qo : Mo;
    v8f acc = {};
#pragma unroll
    for (int ks = 0; ks < 8; ++ks)
      acc = wmma_bf16(load_frag(Asrc, ks, lane), load_frag(w, nt * 8 + ks, lane), acc);
    int n = lane & 15, h = lane >> 4;
    float bias = bb[n0 + n];
#pragma unroll
    for (int r = 0; r < 8; ++r) dst[(r + h * 8) * HD + n0 + n] = acc[r] + bias;
  }
  __syncthreads();

  // one edge per thread: score = q . k[nbr] + bias ; sigmoid gate ; scatter
  {
    int r = tid >> 4, j = tid & 15;
    int n = row0 + r;
    int dstn = nbrs[n * K_ + j];
    const float* kv = kvec + (size_t)dstn * HD;
    float sc = ebias[n * K_ + j];
    for (int d = 0; d < HD; ++d) sc += Qo[r * HD + d] * kv[d];
    float gate = sigmoidf(sc);
    float* dstp = incoming + ((size_t)b * N_ + dstn) * DS;
    for (int d = 0; d < DS; ++d) atomicAdd(&dstp[d], gate * Mo[r * DS + d]);
  }
}

// ---------------- kernel 3: update MLP + decay blend -----------------------
__global__ __launch_bounds__(256) void update_kernel(
    const float* s, const __bf16* cidb, const float* sn, const float* incoming,
    const __bf16* u1w, const float* u1b, const __bf16* u2w, const float* u2b,
    const float* decay, float* out) {
  __shared__ __align__(32) __bf16 X[16 * UW];      // A-frag, 10 ktiles
  __shared__ __align__(32) __bf16 Hh[16 * HID];    // A-frag, 8 ktiles
  int tid = threadIdx.x, wid = tid >> 5, lane = tid & 31;
  int row0 = blockIdx.x * 16;
  int b = blockIdx.y;

  for (int t = tid; t < 16 * UW; t += 256) {
    int m = t / UW, d = t % UW;
    size_t gi = ((size_t)b * N_ + row0 + m) * DS;
    float v;
    if (d < DS)            v = sn[gi + d];
    else if (d < DS + DID) v = (float)cidb[(size_t)(row0 + m) * DID + (d - DS)];
    else                   v = incoming[gi + (d - DS - DID)];
    X[afrag_idx(m, d)] = (__bf16)v;
  }
  __syncthreads();

  // layer1: 320 -> 256, 2 tiles per wave, 10 K-steps
#pragma unroll
  for (int tt = 0; tt < 2; ++tt) {
    int nt = wid * 2 + tt;
    int n0 = nt * 16;
    v8f acc = {};
#pragma unroll
    for (int ks = 0; ks < 10; ++ks)
      acc = wmma_bf16(load_frag(X, ks, lane), load_frag(u1w, nt * 10 + ks, lane), acc);
    int n = lane & 15, h = lane >> 4;
    float bias = u1b[n0 + n];
#pragma unroll
    for (int r = 0; r < 8; ++r)
      Hh[afrag_idx(r + h * 8, n0 + n)] = (__bf16)gelu_tanh(acc[r] + bias);
  }
  __syncthreads();

  // layer2: 256 -> 128, one tile per wave; s_new = decay*s + delta
  {
    int nt = wid;
    int n0 = nt * 16;
    v8f acc = {};
#pragma unroll
    for (int ks = 0; ks < 8; ++ks)
      acc = wmma_bf16(load_frag(Hh, ks, lane), load_frag(u2w, nt * 8 + ks, lane), acc);
    int n = lane & 15, h = lane >> 4;
    float bias = u2b[n0 + n];
#pragma unroll
    for (int r = 0; r < 8; ++r) {
      int m = r + h * 8;
      size_t gi = ((size_t)b * N_ + row0 + m) * DS + n0 + n;
      out[gi] = decay[row0 + m] * s[gi] + acc[r] + bias;
    }
  }
}

// ---------------- host launch ----------------
extern "C" void kernel_launch(void* const* d_in, const int* in_sizes, int n_in,
                              void* d_out, int out_size, void* d_ws, size_t ws_size,
                              hipStream_t stream) {
  const float* s    = (const float*)d_in[0];
  const int*   nbrs = (const int*)d_in[1];
  const float* eb   = (const float*)d_in[2];
  const float* cid  = (const float*)d_in[3];
  const float* cnw  = (const float*)d_in[4];
  const float* c1w  = (const float*)d_in[5];  const float* c1b = (const float*)d_in[6];
  const float* c2w  = (const float*)d_in[7];  const float* c2b = (const float*)d_in[8];
  const float* q1w  = (const float*)d_in[9];  const float* q1b = (const float*)d_in[10];
  const float* q2w  = (const float*)d_in[11]; const float* q2b = (const float*)d_in[12];
  const float* k1w  = (const float*)d_in[13]; const float* k1b = (const float*)d_in[14];
  const float* k2w  = (const float*)d_in[15]; const float* k2b = (const float*)d_in[16];
  const float* u1w  = (const float*)d_in[17]; const float* u1b = (const float*)d_in[18];
  const float* u2w  = (const float*)d_in[19]; const float* u2b = (const float*)d_in[20];
  const float* decw = (const float*)d_in[21]; const float* decb = (const float*)d_in[22];

  char* ws = (char*)d_ws;
  size_t off = 0;
  auto take = [&](size_t bytes) -> char* {
    char* p = ws + off;
    off = (off + bytes + 255) & ~(size_t)255;
    return p;
  };
  __bf16* bc1 = (__bf16*)take((size_t)XW * HID * 2);
  __bf16* bq1 = (__bf16*)take((size_t)XW * HID * 2);
  __bf16* bk1 = (__bf16*)take((size_t)DID * HID * 2);
  __bf16* bu1 = (__bf16*)take((size_t)UW * HID * 2);
  __bf16* bc2 = (__bf16*)take((size_t)HID * HD * 2);
  __bf16* bq2 = (__bf16*)take((size_t)HID * HD * 2);
  __bf16* bk2 = (__bf16*)take((size_t)HID * HD * 2);
  __bf16* bu2 = (__bf16*)take((size_t)HID * HD * 2);
  __bf16* bcid = (__bf16*)take((size_t)N_ * DID * 2);
  float* kvec  = (float*)take((size_t)N_ * HD * 4);
  float* decay = (float*)take((size_t)N_ * 4);
  float* sn    = (float*)take((size_t)B_ * N_ * DS * 4);
  float* inc   = (float*)take((size_t)B_ * N_ * DS * 4);

  hipMemsetAsync(inc, 0, (size_t)B_ * N_ * DS * sizeof(float), stream);

  prep_kernel<<<256, 256, 0, stream>>>(c1w, q1w, k1w, u1w, c2w, q2w, k2w, u2w, cid,
                                       bc1, bq1, bk1, bu1, bc2, bq2, bk2, bu2, bcid);

  kproj_kernel<<<N_ / 16, 256, 0, stream>>>(bcid, cid, bk1, k1b, bk2, k2b,
                                            decw, decb, kvec, decay);

  fused_main_kernel<<<dim3(N_ / 16, B_), 256, 0, stream>>>(
      s, nbrs, eb, bcid, cnw, bc1, c1b, bc2, c2b, bq1, q1b, bq2, q2b,
      kvec, sn, inc);

  update_kernel<<<dim3(N_ / 16, B_), 256, 0, stream>>>(
      s, bcid, sn, inc, bu1, u1b, bu2, u2b, decay, (float*)d_out);
}
